// ReED_80315888435553
// MI455X (gfx1250) — matrix operand order repaired and checked by reading, not compiled, not run on hardware
//
#include <hip/hip_runtime.h>

#define NUM_ENT 10000
#define NUM_REL 50
#define HID     64
#define N_TRIP  500000
#define NBIN    (NUM_REL * NUM_ENT)   // (r,h) counting-sort bins = 500000
#define TILES   31300                  // ceil bound: (500000 + 50*16)/16
#define IDXCAP  (TILES * 16)           // 500800 padded slots
#define NTILE_B 625                    // NUM_ENT/16 base-GEMM tiles
#define TPB     8                      // tiles per block in WMMA kernels

typedef unsigned short u16;
typedef unsigned int   u32;
typedef unsigned long long u64;
typedef __bf16 v16bf __attribute__((ext_vector_type(16)));
typedef float  v8f   __attribute__((ext_vector_type(8)));

struct alignas(16) QW   { u64 lo, hi; };   // 16-byte move unit
struct alignas(16) Frag { QW q0, q1; };    // 32 bytes == v16bf

__device__ __forceinline__ u16 f2bf(float f) { // RNE f32->bf16
  u32 u = __float_as_uint(f);
  u32 r = u + 0x7FFFu + ((u >> 16) & 1u);
  return (u16)(r >> 16);
}

__device__ __forceinline__ v8f wmma_bf16(v16bf a, v16bf b, v8f c) {
  return __builtin_amdgcn_wmma_f32_16x16x32_bf16(false, a, false, b, (short)0, c,
                                                 false, false);
}

// ---------------- init ----------------

__global__ void k_init_ent(const float* __restrict__ Wres0, float* __restrict__ entF) {
  int i = blockIdx.x * blockDim.x + threadIdx.x;
  if (i >= NUM_ENT * HID) return;
  int e = i >> 6, j = i & 63;
  entF[i] = Wres0[j * NUM_ENT + e];             // Wres0^T
}

__global__ void k_init_rel(const float* __restrict__ Wpr0, float* __restrict__ rel1) {
  int i = blockIdx.x * blockDim.x + threadIdx.x;
  if (i >= NUM_REL * HID) return;
  int rr = i >> 6, j = i & 63;
  rel1[i] = Wpr0[j * NUM_REL + rr];             // Wpr0^T
}

__global__ void k_fill(int* __restrict__ hS, int* __restrict__ counts2) {
  int i = blockIdx.x * blockDim.x + threadIdx.x;
  if (i < IDXCAP) hS[i] = -1;                   // sentinel for padded slots
  if (i < NBIN)   counts2[i] = 0;
}

// ------------- (r,h) counting sort, r-groups padded to x16 -------------

__global__ void k_hist2(const int* __restrict__ h, const int* __restrict__ r,
                        int* __restrict__ counts2) {
  int i = blockIdx.x * blockDim.x + threadIdx.x;
  if (i >= N_TRIP) return;
  atomicAdd(&counts2[r[i] * NUM_ENT + h[i]], 1);
}

// 50 blocks x 256: exclusive scan of each relation's 10000 h-bins
__global__ void k_scan(const int* __restrict__ counts2, int* __restrict__ cursor2,
                       int* __restrict__ gtot) {
  __shared__ int ssum[256];
  int g = blockIdx.x, tid = threadIdx.x;
  int base = g * NUM_ENT;
  int start = tid * 40;
  int end = start + 40; if (end > NUM_ENT) end = NUM_ENT;
  int s = 0;
  for (int k = start; k < end; ++k) s += counts2[base + k];
  ssum[tid] = s;
  __syncthreads();
  if (tid == 0) {
    int run = 0;
    for (int i = 0; i < 256; ++i) { int v = ssum[i]; ssum[i] = run; run += v; }
    gtot[g] = run;
  }
  __syncthreads();
  int off = ssum[tid];
  for (int k = start; k < end; ++k) { cursor2[base + k] = off; off += counts2[base + k]; }
}

__global__ void k_gbase(const int* __restrict__ gtot, int* __restrict__ gbase) {
  if (blockIdx.x == 0 && threadIdx.x == 0) {
    int off = 0;
    for (int g = 0; g < NUM_REL; ++g) {
      gbase[g] = off;
      off += (gtot[g] + 15) & ~15;              // pad each r-group to x16
    }
  }
}

__global__ void k_sort2(const int* __restrict__ h, const int* __restrict__ r,
                        const int* __restrict__ t, const int* __restrict__ gbase,
                        int* __restrict__ cursor2, int* __restrict__ hS,
                        int* __restrict__ rS, int* __restrict__ tS) {
  int i = blockIdx.x * blockDim.x + threadIdx.x;
  if (i >= N_TRIP) return;
  int hh = h[i], rr = r[i], tt = t[i];
  int pos = gbase[rr] + atomicAdd(&cursor2[rr * NUM_ENT + hh], 1);
  hS[pos] = hh; rS[pos] = rr; tS[pos] = tt;
}

// -------- layer 0 scatter: h-sorted lanes => coalesced Wme0 rows --------

__global__ void k_l0_scatter(const float* __restrict__ Wme0,
                             const float* __restrict__ Wmr0,
                             const int* __restrict__ hS, const int* __restrict__ rS,
                             const int* __restrict__ tS, float* __restrict__ entF) {
  int p = blockIdx.x * blockDim.x + threadIdx.x;
  if (p >= IDXCAP) return;
  int hh = hS[p];
  if (hh < 0) return;                           // padding slot
  int rr = rS[p], tt = tS[p];
  int jb = blockIdx.y * 8;
  #pragma unroll
  for (int jj = 0; jj < 8; ++jj) {
    int j = jb + jj;
    int row = rr * HID + j;
    float m = Wme0[(size_t)row * NUM_ENT + hh] + Wmr0[row * NUM_REL + rr];
    atomicAdd(&entF[tt * HID + j], m);
  }
}

__global__ void k_act0(float* __restrict__ entF, u16* __restrict__ entB) {
  int i = blockIdx.x * blockDim.x + threadIdx.x;
  if (i >= NUM_ENT * HID) return;
  float f = entF[i];
  float l = f > 0.0f ? f : 0.01f * f;
  entF[i] = l;
  entB[i] = f2bf(l);
}

// ---------------- layer-1 weight prep ----------------

__global__ void k_cvt(const float* __restrict__ Wme1, const float* __restrict__ Wres1,
                      u16* __restrict__ wme1b, u16* __restrict__ wres1b) {
  int i = blockIdx.x * blockDim.x + threadIdx.x;
  if (i < NUM_REL * HID * HID) wme1b[i] = f2bf(Wme1[i]);
  if (i < HID * HID)           wres1b[i] = f2bf(Wres1[i]);
}

__global__ void k_diag(const float* __restrict__ Wmr1, const float* __restrict__ rel1,
                       float* __restrict__ diag1) {
  int i = blockIdx.x * blockDim.x + threadIdx.x;   // i = r*64 + j
  if (i >= NUM_REL * HID) return;
  int rr = i >> 6;
  float s = 0.0f;
  for (int k = 0; k < HID; ++k) s += Wmr1[i * HID + k] * rel1[rr * HID + k];
  diag1[i] = s;
}

// ------- WMMA base GEMM: upd1 = emb_ent1 @ Wres1^T (B staged once) -------

__global__ void __launch_bounds__(32)
k_base_wmma(const u16* __restrict__ entB, const u16* __restrict__ wres1b,
            float* __restrict__ outEnt) {
  __shared__ __align__(16) u16 sA[16 * 64];
  __shared__ __align__(16) u16 sB[64 * 64];
  int lane = threadIdx.x;
  int mrow = lane & 15, ksel = lane >> 4, msel = ksel * 8;

  const QW* bsrc = (const QW*)wres1b;
  QW* bdst = (QW*)sB;
  #pragma unroll
  for (int it = 0; it < 16; ++it) bdst[lane + it * 32] = bsrc[lane + it * 32];

  for (int sub = 0; sub < TPB; ++sub) {
    int tile = blockIdx.x * TPB + sub;
    if (tile >= NTILE_B) break;

    const QW* asrc = (const QW*)(entB + (size_t)tile * 16 * 64);
    QW* adst = (QW*)sA;
    #pragma unroll
    for (int it = 0; it < 4; ++it) adst[lane + it * 32] = asrc[lane + it * 32];
    __syncthreads();

    const QW* arow = (const QW*)(sA + mrow * 64);
    Frag a0, a1;
    a0.q0 = arow[ksel];     a0.q1 = arow[2 + ksel];   // K 0..31
    a1.q0 = arow[4 + ksel]; a1.q1 = arow[6 + ksel];   // K 32..63
    v16bf av0 = __builtin_bit_cast(v16bf, a0);
    v16bf av1 = __builtin_bit_cast(v16bf, a1);

    v8f acc[4];
    #pragma unroll
    for (int nt = 0; nt < 4; ++nt) {
      int n = mrow + nt * 16;
      const QW* brow = (const QW*)(sB + n * 64);
      Frag b0, b1;
      b0.q0 = brow[ksel];     b0.q1 = brow[2 + ksel];
      b1.q0 = brow[4 + ksel]; b1.q1 = brow[6 + ksel];
      v8f c = {};
      c = wmma_bf16(av0, __builtin_bit_cast(v16bf, b0), c);
      c = wmma_bf16(av1, __builtin_bit_cast(v16bf, b1), c);
      acc[nt] = c;
    }

    #pragma unroll
    for (int nt = 0; nt < 4; ++nt)
      #pragma unroll
      for (int g = 0; g < 8; ++g) {
        int row = tile * 16 + msel + g;
        outEnt[(size_t)row * HID + mrow + nt * 16] = acc[nt][g];
      }
    __syncthreads();                 // protect sA reuse next tile
  }
}

// -- WMMA grouped GEMM + scatter; B/diag re-staged only on relation change --

__global__ void __launch_bounds__(32)
k_msg_wmma(const int* __restrict__ hS, const int* __restrict__ rS,
           const int* __restrict__ tS, const u16* __restrict__ entB,
           const u16* __restrict__ wme1b, const float* __restrict__ diag1,
           float* __restrict__ outEnt) {
  __shared__ __align__(16) u16 sA[16 * 64];
  __shared__ __align__(16) u16 sB[64 * 64];
  __shared__ __align__(16) float sDiag[64];
  __shared__ int sH[16];
  __shared__ int sT[16];

  int lane = threadIdx.x;
  int mrow = lane & 15, ksel = lane >> 4, msel = ksel * 8;
  int curR = -1;

  for (int sub = 0; sub < TPB; ++sub) {
    int tile = blockIdx.x * TPB + sub;
    if (tile >= TILES) break;
    int p0 = tile * 16;
    int h0 = hS[p0];                 // uniform across wave
    if (h0 < 0) continue;            // unused tile past padded total
    int rr = rS[p0];                 // whole tile shares relation rr

    if (rr != curR) {                // uniform branch: stage 8KB weight block
      curR = rr;
      const QW* bsrc = (const QW*)(wme1b + (size_t)rr * HID * HID);
      QW* bdst = (QW*)sB;
      #pragma unroll
      for (int it = 0; it < 16; ++it) bdst[lane + it * 32] = bsrc[lane + it * 32];
      if (lane < 16)
        ((QW*)sDiag)[lane] = ((const QW*)(diag1 + rr * HID))[lane];
    }
    if (lane < 16) {
      int hh = hS[p0 + lane];
      sH[lane] = hh;
      sT[lane] = (hh >= 0) ? tS[p0 + lane] : -1;
    }
    __syncthreads();

    // stage A: 16 gathered ent rows (h-sorted => L2-local), sentinels -> 0
    #pragma unroll
    for (int it = 0; it < 4; ++it) {
      int q = lane + it * 32;        // 0..127 16B chunks
      int row = q >> 3, c = q & 7;
      int hh = sH[row];
      QW v; v.lo = 0; v.hi = 0;
      if (hh >= 0) v = *(const QW*)(entB + (size_t)hh * HID + c * 8);
      *(QW*)(sA + row * 64 + c * 8) = v;
    }
    __syncthreads();

    const QW* arow = (const QW*)(sA + mrow * 64);
    Frag a0, a1;
    a0.q0 = arow[ksel];     a0.q1 = arow[2 + ksel];
    a1.q0 = arow[4 + ksel]; a1.q1 = arow[6 + ksel];
    v16bf av0 = __builtin_bit_cast(v16bf, a0);
    v16bf av1 = __builtin_bit_cast(v16bf, a1);

    v8f acc[4];
    #pragma unroll
    for (int nt = 0; nt < 4; ++nt) {
      int n = mrow + nt * 16;
      const QW* brow = (const QW*)(sB + n * 64);
      Frag b0, b1;
      b0.q0 = brow[ksel];     b0.q1 = brow[2 + ksel];
      b1.q0 = brow[4 + ksel]; b1.q1 = brow[6 + ksel];
      v8f c = {};
      c = wmma_bf16(av0, __builtin_bit_cast(v16bf, b0), c);
      c = wmma_bf16(av1, __builtin_bit_cast(v16bf, b1), c);
      acc[nt] = c;
    }

    #pragma unroll
    for (int nt = 0; nt < 4; ++nt)
      #pragma unroll
      for (int g = 0; g < 8; ++g) {
        int m = msel + g;
        int tt = sT[m];
        if (tt >= 0) {
          int j = mrow + nt * 16;
          atomicAdd(&outEnt[(size_t)tt * HID + j], acc[nt][g] + sDiag[j]);
        }
      }
    __syncthreads();                 // protect sA/sH/sT reuse next tile
  }
}

// ---------------- epilogue ----------------

__global__ void k_act_out(float* __restrict__ outEnt) {
  int i = blockIdx.x * blockDim.x + threadIdx.x;
  if (i >= NUM_ENT * HID) return;
  float f = outEnt[i];
  outEnt[i] = f > 0.0f ? f : 0.01f * f;
}

__global__ void k_rel_out(const float* __restrict__ rel1, const float* __restrict__ Wpr1,
                          float* __restrict__ outRel) {
  int i = blockIdx.x * blockDim.x + threadIdx.x;   // i = r*64 + j
  if (i >= NUM_REL * HID) return;
  int rr = i >> 6, j = i & 63;
  float s = 0.0f;
  for (int k = 0; k < HID; ++k) s += rel1[rr * HID + k] * Wpr1[j * HID + k];
  outRel[i] = s;
}

// ---------------- host launcher ----------------

extern "C" void kernel_launch(void* const* d_in, const int* in_sizes, int n_in,
                              void* d_out, int out_size, void* d_ws, size_t ws_size,
                              hipStream_t stream) {
  const float* Wres0 = (const float*)d_in[0];
  const float* Wme0  = (const float*)d_in[1];
  const float* Wmr0  = (const float*)d_in[2];
  const float* Wpr0  = (const float*)d_in[3];
  const float* Wres1 = (const float*)d_in[4];
  const float* Wme1  = (const float*)d_in[5];
  const float* Wmr1  = (const float*)d_in[6];
  const float* Wpr1  = (const float*)d_in[7];
  const int* h = (const int*)d_in[8];
  const int* r = (const int*)d_in[9];
  const int* t = (const int*)d_in[10];

  float* outEnt = (float*)d_out;
  float* outRel = outEnt + (size_t)NUM_ENT * HID;

  char* ws = (char*)d_ws;
  size_t o = 0;
  float* entF    = (float*)(ws + o); o += (size_t)NUM_ENT * HID * 4;       // 2.56 MB
  u16*   entB    = (u16*)  (ws + o); o += (size_t)NUM_ENT * HID * 2;       // 1.28 MB
  float* rel1    = (float*)(ws + o); o += (size_t)NUM_REL * HID * 4;
  float* diag1   = (float*)(ws + o); o += (size_t)NUM_REL * HID * 4;
  u16*   wme1b   = (u16*)  (ws + o); o += (size_t)NUM_REL * HID * HID * 2; // 400 KB
  u16*   wres1b  = (u16*)  (ws + o); o += (size_t)HID * HID * 2;
  int*   counts2 = (int*)  (ws + o); o += (size_t)NBIN * 4;                // 2 MB
  int*   cursor2 = (int*)  (ws + o); o += (size_t)NBIN * 4;                // 2 MB
  int*   gtot    = (int*)  (ws + o); o += 64 * 4;
  int*   gbase   = (int*)  (ws + o); o += 64 * 4;
  int*   hS      = (int*)  (ws + o); o += (size_t)IDXCAP * 4;              // 2 MB
  int*   rS      = (int*)  (ws + o); o += (size_t)IDXCAP * 4;
  int*   tS      = (int*)  (ws + o); o += (size_t)IDXCAP * 4;

  const int T = 256;
  // init + (r,h) counting sort with per-relation padding
  k_init_ent<<<(NUM_ENT * HID + T - 1) / T, T, 0, stream>>>(Wres0, entF);
  k_init_rel<<<(NUM_REL * HID + T - 1) / T, T, 0, stream>>>(Wpr0, rel1);
  k_fill<<<(IDXCAP + T - 1) / T, T, 0, stream>>>(hS, counts2);
  k_hist2<<<(N_TRIP + T - 1) / T, T, 0, stream>>>(h, r, counts2);
  k_scan<<<NUM_REL, 256, 0, stream>>>(counts2, cursor2, gtot);
  k_gbase<<<1, 32, 0, stream>>>(gtot, gbase);
  k_sort2<<<(N_TRIP + T - 1) / T, T, 0, stream>>>(h, r, t, gbase, cursor2, hS, rS, tS);
  // layer 0 (coalesced gather/scatter) + activation
  k_l0_scatter<<<dim3((IDXCAP + T - 1) / T, 8), T, 0, stream>>>(Wme0, Wmr0, hS, rS, tS, entF);
  k_act0<<<(NUM_ENT * HID + T - 1) / T, T, 0, stream>>>(entF, entB);
  // layer-1 weight prep
  k_cvt<<<(NUM_REL * HID * HID + T - 1) / T, T, 0, stream>>>(Wme1, Wres1, wme1b, wres1b);
  k_diag<<<(NUM_REL * HID + T - 1) / T, T, 0, stream>>>(Wmr1, rel1, diag1);
  // WMMA GEMMs
  k_base_wmma<<<(NTILE_B + TPB - 1) / TPB, 32, 0, stream>>>(entB, wres1b, outEnt);
  k_msg_wmma<<<(TILES + TPB - 1) / TPB, 32, 0, stream>>>(hS, rS, tS, entB, wme1b, diag1, outEnt);
  // epilogue
  k_act_out<<<(NUM_ENT * HID + T - 1) / T, T, 0, stream>>>(outEnt);
  k_rel_out<<<(NUM_REL * HID + T - 1) / T, T, 0, stream>>>(rel1, Wpr1, outRel);
}
